// DDIModel_18803366822360
// MI455X (gfx1250) — compile-verified, matrix operand order checked by reading, not run on hardware
//
#include <hip/hip_runtime.h>
#include <hip/hip_bf16.h>

// ---------------------------------------------------------------------------
// GATv2 x3 on gfx1250 (MI455X).
//  - Dense GEMMs: V_WMMA_F32_16X16X32_BF16; A-strip staged to LDS via the
//    Tensor Data Mover (tensor_load_to_lds + s_wait_tensorcnt), 4 output
//    tiles per wave reusing one A fragment.
//  - Sparse attention: wave32 edge-parallel kernels with f32 L2 atomics.
// ---------------------------------------------------------------------------

#define NNODES 50000
#define FIN    256
#define HID    512
#define HEADS  8
#define CPH    64      // channels per head

typedef __bf16        v16bf __attribute__((ext_vector_type(16)));
typedef float         v8f   __attribute__((ext_vector_type(8)));
typedef unsigned int  v4u   __attribute__((ext_vector_type(4)));
typedef unsigned int  u32x4 __attribute__((ext_vector_type(4)));
typedef int           i32x4 __attribute__((ext_vector_type(4)));
typedef int           i32x8 __attribute__((ext_vector_type(8)));

union ABFrag { v4u q[2]; v16bf v; };

// ---- fp32 -> bf16 (round to nearest even) ----------------------------------
__device__ __forceinline__ unsigned short f2bf(float f) {
    unsigned u = __float_as_uint(f);
    unsigned r = ((u >> 16) & 1u) + 0x7fffu;
    return (unsigned short)((u + r) >> 16);
}

__global__ void f32_to_bf16_kernel(const float* __restrict__ in,
                                   unsigned short* __restrict__ out, size_t n) {
    size_t i = (size_t)blockIdx.x * blockDim.x + threadIdx.x;
    if (i < n) out[i] = f2bf(in[i]);
}

// W [K x 512] fp32 (row-major) -> Wt [512 x K] bf16 (row-major)
__global__ void w_transpose_bf16_kernel(const float* __restrict__ W,
                                        unsigned short* __restrict__ Wt, int K) {
    size_t i = (size_t)blockIdx.x * blockDim.x + threadIdx.x;
    size_t total = (size_t)HID * K;
    if (i >= total) return;
    int m = (int)(i / K);
    int k = (int)(i % K);
    Wt[(size_t)m * K + k] = f2bf(W[(size_t)k * HID + m]);
}

__global__ void fill_f32_kernel(float* __restrict__ p, float v, size_t n) {
    size_t i = (size_t)blockIdx.x * blockDim.x + threadIdx.x;
    if (i < n) p[i] = v;
}

// ---------------------------------------------------------------------------
// TDM: stage a 16 x K bf16 strip (row stride K) from global into LDS offset 0.
// D# built per CDNA5 ISA (2D tile, data_size = 4 bytes).
// ---------------------------------------------------------------------------
__device__ __forceinline__ void tdm_load_strip(const unsigned short* gsrc, int K) {
#if __has_builtin(__builtin_amdgcn_tensor_load_to_lds)
    const unsigned long long ga = (unsigned long long)(const void*)gsrc;
    const unsigned d0 = (unsigned)(K / 2);       // row length in 4B units

    u32x4 g0;
    g0[0] = 1u;                                   // count = 1 valid descriptor
    g0[1] = 0u;                                   // lds_addr = 0 (single LDS array)
    g0[2] = (unsigned)(ga & 0xFFFFFFFFull);       // global_addr[31:0]
    g0[3] = (unsigned)((ga >> 32) & 0x01FFFFFFull) | (2u << 30);  // addr[56:32] | type=2

    i32x8 g1;
    g1[0] = (int)(2u << 16);                      // workgroup_mask=0, data_size=2 (4B)
    g1[1] = (int)((d0 & 0xFFFFu) << 16);          // tensor_dim0[15:0] at bits 63:48
    g1[2] = (int)((d0 >> 16) | (16u << 16));      // tensor_dim0[31:16] | tensor_dim1=16
    g1[3] = (int)((d0 & 0xFFFFu) << 16);          // tensor_dim1 hi=0 | tile_dim0 = d0
    g1[4] = 16;                                   // tile_dim1 = 16 rows, tile_dim2 = 0
    g1[5] = (int)d0;                              // tensor_dim0_stride[31:0] = row stride
    g1[6] = 0;                                    // stride hi / dim1_stride lo
    g1[7] = 0;

    i32x4 g2 = {0, 0, 0, 0};
    i32x4 g3 = {0, 0, 0, 0};

#if defined(__clang_major__) && (__clang_major__ >= 23)
    i32x8 g2b = {0, 0, 0, 0, 0, 0, 0, 0};
    __builtin_amdgcn_tensor_load_to_lds(g0, g1, g2, g3, g2b, 0);
#else
    __builtin_amdgcn_tensor_load_to_lds(g0, g1, g2, g3, 0);
#endif
    __builtin_amdgcn_s_wait_tensorcnt(0);
#else
    (void)gsrc; (void)K;
#endif
}

// ---------------------------------------------------------------------------
// GEMM: out[N x 512] = xb[N x K](bf16) @ Wt^T (Wt is [512 x K] bf16) + bias
// Block = 256 threads (8 waves) -> full 16 x 512 output strip.
// A strip staged in LDS by TDM; wave w computes 4 tiles (cols w*64..w*64+63),
// reusing one A fragment per K-step (4 v_wmma per A ds-load).
// grid = (N/16).
// ---------------------------------------------------------------------------
__global__ __launch_bounds__(256)
void gemm_wmma_bf16_kernel(const unsigned short* __restrict__ xb,
                           const unsigned short* __restrict__ wt,
                           const float* __restrict__ bias,
                           float* __restrict__ out, int K) {
    __shared__ unsigned short tileA[16 * HID];   // 16 KB max (K <= 512)

    const int lane    = threadIdx.x & 31;
    const int wave    = threadIdx.x >> 5;
    const int rowBase = blockIdx.x * 16;
    const int colBase = wave * 64;

    const unsigned short* gsrc = xb + (size_t)rowBase * K;

#if __has_builtin(__builtin_amdgcn_tensor_load_to_lds)
    if (threadIdx.x < 32) tdm_load_strip(gsrc, K);
#else
    {   // cooperative fallback copy (16B chunks)
        const v4u* s = (const v4u*)gsrc;
        v4u*       d = (v4u*)tileA;
        for (int i = threadIdx.x; i < 2 * K; i += 256) d[i] = s[i];
    }
#endif
    __syncthreads();

    const int khalf = (lane >> 4) * 8;           // 0 or 8

    const unsigned short* pa  = tileA + (lane & 15) * K + khalf;          // LDS
    const unsigned short* pb0 = wt + (size_t)(colBase +  0 + (lane & 15)) * K + khalf;
    const unsigned short* pb1 = pb0 + (size_t)16 * K;
    const unsigned short* pb2 = pb0 + (size_t)32 * K;
    const unsigned short* pb3 = pb0 + (size_t)48 * K;

    v8f c0 = {}, c1 = {}, c2 = {}, c3 = {};
    for (int kk = 0; kk < K; kk += 32) {
        __builtin_prefetch(pb0 + 32, 0, 1);      // global_prefetch_b8
        __builtin_prefetch(pb1 + 32, 0, 1);
        __builtin_prefetch(pb2 + 32, 0, 1);
        __builtin_prefetch(pb3 + 32, 0, 1);
        ABFrag A, B0, B1, B2, B3;
        A.q[0]  = *(const v4u*)(pa);             // ds_load_b128
        A.q[1]  = *(const v4u*)(pa + 16);
        B0.q[0] = *(const v4u*)(pb0); B0.q[1] = *(const v4u*)(pb0 + 16);
        B1.q[0] = *(const v4u*)(pb1); B1.q[1] = *(const v4u*)(pb1 + 16);
        B2.q[0] = *(const v4u*)(pb2); B2.q[1] = *(const v4u*)(pb2 + 16);
        B3.q[0] = *(const v4u*)(pb3); B3.q[1] = *(const v4u*)(pb3 + 16);
        c0 = __builtin_amdgcn_wmma_f32_16x16x32_bf16(false, A.v, false, B0.v, (short)0, c0, false, false);
        c1 = __builtin_amdgcn_wmma_f32_16x16x32_bf16(false, A.v, false, B1.v, (short)0, c1, false, false);
        c2 = __builtin_amdgcn_wmma_f32_16x16x32_bf16(false, A.v, false, B2.v, (short)0, c2, false, false);
        c3 = __builtin_amdgcn_wmma_f32_16x16x32_bf16(false, A.v, false, B3.v, (short)0, c3, false, false);
        pa += 32; pb0 += 32; pb1 += 32; pb2 += 32; pb3 += 32;
    }

    const int rbase = rowBase + (lane >> 4) * 8;
    v8f accs[4] = {c0, c1, c2, c3};
    #pragma unroll
    for (int t = 0; t < 4; ++t) {
        const int colg = colBase + t * 16 + (lane & 15);
        const float bv = bias[colg];
        #pragma unroll
        for (int r = 0; r < 8; ++r)
            out[(size_t)(rbase + r) * HID + colg] = accs[t][r] + bv;
    }
}

// ---------------------------------------------------------------------------
// Ordered-float atomic max (init to -inf): int-max for >=0, uint-min for <0.
// ---------------------------------------------------------------------------
__device__ __forceinline__ void atomicMaxF(float* addr, float v) {
    if (v >= 0.0f) atomicMax((int*)addr, __float_as_int(v));
    else           atomicMin((unsigned int*)addr, __float_as_uint(v));
}

__device__ __forceinline__ float waveSum(float v) {
    #pragma unroll
    for (int o = 16; o > 0; o >>= 1) v += __shfl_xor(v, o, 32);
    return v;
}

// ---------------------------------------------------------------------------
// Pass 1: per edge, per head: logit = att[h] . leaky_relu(xl[src]+xr[dst])
// One wave32 per edge; lane owns 16 channels (4 lanes per head).
// ---------------------------------------------------------------------------
__global__ __launch_bounds__(256)
void edge_logits_kernel(const float* __restrict__ xl, const float* __restrict__ xr,
                        const float* __restrict__ att,
                        const int* __restrict__ srcIdx, const int* __restrict__ dstIdx,
                        float* __restrict__ logits, float* __restrict__ m,
                        int E, int Etot) {
    int e = blockIdx.x * 8 + (threadIdx.x >> 5);
    if (e >= Etot) return;
    const int lane = threadIdx.x & 31;
    const int src  = (e < E) ? srcIdx[e] : (e - E);
    const int dst  = (e < E) ? dstIdx[e] : (e - E);
    const int h    = lane >> 2;

    const float4* pl = (const float4*)(xl + (size_t)src * HID + lane * 16);
    const float4* pr = (const float4*)(xr + (size_t)dst * HID + lane * 16);
    const float4* pa = (const float4*)(att + h * CPH + (lane & 3) * 16);

    float s = 0.0f;
    #pragma unroll
    for (int q = 0; q < 4; ++q) {
        float4 a = pl[q], b = pr[q], w = pa[q];
        float v0 = a.x + b.x; v0 = v0 > 0.f ? v0 : 0.2f * v0;
        float v1 = a.y + b.y; v1 = v1 > 0.f ? v1 : 0.2f * v1;
        float v2 = a.z + b.z; v2 = v2 > 0.f ? v2 : 0.2f * v2;
        float v3 = a.w + b.w; v3 = v3 > 0.f ? v3 : 0.2f * v3;
        s += w.x * v0 + w.y * v1 + w.z * v2 + w.w * v3;
    }
    s += __shfl_xor(s, 1, 32);
    s += __shfl_xor(s, 2, 32);
    if ((lane & 3) == 0) {
        logits[(size_t)e * HEADS + h] = s;
        atomicMaxF(&m[(size_t)dst * HEADS + h], s);
    }
}

// Pass 2: alpha = exp(logit - m[dst]); denom[dst] += alpha
__global__ void edge_alpha_kernel(const float* __restrict__ logits,
                                  const float* __restrict__ m,
                                  const int* __restrict__ dstIdx,
                                  float* __restrict__ alpha, float* __restrict__ denom,
                                  int E, int Etot) {
    size_t idx = (size_t)blockIdx.x * blockDim.x + threadIdx.x;
    if (idx >= (size_t)Etot * HEADS) return;
    int e = (int)(idx >> 3);
    int h = (int)(idx & 7);
    int dst = (e < E) ? dstIdx[e] : (e - E);
    float a = __expf(logits[idx] - m[(size_t)dst * HEADS + h]);
    alpha[idx] = a;
    atomicAdd(&denom[(size_t)dst * HEADS + h], a);
}

// Pass 3: agg[dst] += (alpha/denom[dst]) * xl[src]   (wave per edge)
__global__ __launch_bounds__(256)
void edge_aggregate_kernel(const float* __restrict__ xl,
                           const float* __restrict__ alpha, const float* __restrict__ denom,
                           const int* __restrict__ srcIdx, const int* __restrict__ dstIdx,
                           float* __restrict__ agg, int E, int Etot) {
    int e = blockIdx.x * 8 + (threadIdx.x >> 5);
    if (e >= Etot) return;
    const int lane = threadIdx.x & 31;
    const int src  = (e < E) ? srcIdx[e] : (e - E);
    const int dst  = (e < E) ? dstIdx[e] : (e - E);
    const int h    = lane >> 2;

    const float w = alpha[(size_t)e * HEADS + h] / denom[(size_t)dst * HEADS + h];
    const float* ps = xl  + (size_t)src * HID + lane * 16;
    float*       pd = agg + (size_t)dst * HID + lane * 16;
    #pragma unroll
    for (int i = 0; i < 16; ++i)
        atomicAdd(&pd[i], w * ps[i]);
}

// Fused: y = elu(LN(agg + bias)) (+ residual); one wave32 per node.
__global__ __launch_bounds__(256)
void post_kernel(const float* __restrict__ agg, const float* __restrict__ bias,
                 const float* __restrict__ g, const float* __restrict__ b,
                 const float* __restrict__ residual, float* __restrict__ out, int n) {
    int node = blockIdx.x * 8 + (threadIdx.x >> 5);
    if (node >= n) return;
    const int lane = threadIdx.x & 31;
    const size_t base = (size_t)node * HID + lane * 16;

    float v[16];
    float s = 0.0f;
    #pragma unroll
    for (int i = 0; i < 16; ++i) {
        v[i] = agg[base + i] + bias[lane * 16 + i];
        s += v[i];
    }
    const float mu = waveSum(s) * (1.0f / HID);
    float s2 = 0.0f;
    #pragma unroll
    for (int i = 0; i < 16; ++i) { float d = v[i] - mu; s2 += d * d; }
    const float rs = rsqrtf(waveSum(s2) * (1.0f / HID) + 1e-5f);

    #pragma unroll
    for (int i = 0; i < 16; ++i) {
        int ch = lane * 16 + i;
        float y = (v[i] - mu) * rs * g[ch] + b[ch];
        y = y > 0.0f ? y : expm1f(y);
        if (residual) y += residual[base + i];
        out[base + i] = y;
    }
}

// ---------------------------------------------------------------------------
extern "C" void kernel_launch(void* const* d_in, const int* in_sizes, int n_in,
                              void* d_out, int out_size, void* d_ws, size_t ws_size,
                              hipStream_t stream) {
    const int N    = NNODES;
    const int E    = in_sizes[1] / 2;
    const int Etot = E + N;

    const float* x   = (const float*)d_in[0];
    const int*   ei  = (const int*)d_in[1];
    const int*   src = ei;
    const int*   dst = ei + E;

    // ---- carve workspace ----
    char* ws = (char*)d_ws;
    auto carve = [&](size_t bytes) -> char* {
        char* p = ws;
        ws += (bytes + 255) & ~(size_t)255;
        return p;
    };
    unsigned short* xb  = (unsigned short*)carve((size_t)N * HID * 2);
    unsigned short* Wlt = (unsigned short*)carve((size_t)HID * HID * 2);
    unsigned short* Wrt = (unsigned short*)carve((size_t)HID * HID * 2);
    float* xl     = (float*)carve((size_t)N * HID * 4);
    float* xr     = (float*)carve((size_t)N * HID * 4);
    float* logits = (float*)carve((size_t)Etot * HEADS * 4);
    float* alpha  = (float*)carve((size_t)Etot * HEADS * 4);
    float* mmax   = (float*)carve((size_t)N * HEADS * 4);
    float* denom  = (float*)carve((size_t)N * HEADS * 4);
    float* agg    = (float*)carve((size_t)N * HID * 4);
    float* x1buf  = (float*)carve((size_t)N * HID * 4);
    float* x2buf  = (float*)carve((size_t)N * HID * 4);

    const float NEG_INF = -__builtin_inff();

    auto run_layer = [&](const float* xin, int K, int base,
                         const float* residual, float* xout) {
        const float* Wl   = (const float*)d_in[base + 0];
        const float* Wr   = (const float*)d_in[base + 1];
        const float* bl   = (const float*)d_in[base + 2];
        const float* br   = (const float*)d_in[base + 3];
        const float* att  = (const float*)d_in[base + 4];
        const float* bias = (const float*)d_in[base + 5];
        const float* g    = (const float*)d_in[base + 6];
        const float* b    = (const float*)d_in[base + 7];

        size_t nK = (size_t)N * K;
        f32_to_bf16_kernel<<<(unsigned)((nK + 255) / 256), 256, 0, stream>>>(xin, xb, nK);
        size_t wK = (size_t)HID * K;
        w_transpose_bf16_kernel<<<(unsigned)((wK + 255) / 256), 256, 0, stream>>>(Wl, Wlt, K);
        w_transpose_bf16_kernel<<<(unsigned)((wK + 255) / 256), 256, 0, stream>>>(Wr, Wrt, K);

        gemm_wmma_bf16_kernel<<<N / 16, 256, 0, stream>>>(xb, Wlt, bl, xl, K);
        gemm_wmma_bf16_kernel<<<N / 16, 256, 0, stream>>>(xb, Wrt, br, xr, K);

        size_t nh = (size_t)N * HEADS;
        fill_f32_kernel<<<(unsigned)((nh + 255) / 256), 256, 0, stream>>>(mmax, NEG_INF, nh);
        fill_f32_kernel<<<(unsigned)((nh + 255) / 256), 256, 0, stream>>>(denom, 0.0f, nh);
        size_t nhid = (size_t)N * HID;
        fill_f32_kernel<<<(unsigned)((nhid + 255) / 256), 256, 0, stream>>>(agg, 0.0f, nhid);

        unsigned eb = (unsigned)((Etot + 7) / 8);
        edge_logits_kernel<<<eb, 256, 0, stream>>>(xl, xr, att, src, dst,
                                                   logits, mmax, E, Etot);
        size_t eh = (size_t)Etot * HEADS;
        edge_alpha_kernel<<<(unsigned)((eh + 255) / 256), 256, 0, stream>>>(
            logits, mmax, dst, alpha, denom, E, Etot);
        edge_aggregate_kernel<<<eb, 256, 0, stream>>>(xl, alpha, denom, src, dst,
                                                      agg, E, Etot);
        post_kernel<<<(unsigned)((N + 7) / 8), 256, 0, stream>>>(
            agg, bias, g, b, residual, xout, N);
    };

    run_layer(x,     FIN, 2,  nullptr, x1buf);
    run_layer(x1buf, HID, 10, x1buf,   x2buf);
    run_layer(x2buf, HID, 18, x2buf,   (float*)d_out);
}
